// GatingRe2blocks_34402688041560
// MI455X (gfx1250) — matrix operand order repaired and checked by reading, not compiled, not run on hardware
//
#include <hip/hip_runtime.h>
#include <hip/hip_bf16.h>
#include <math.h>

#define EPSV 1e-5f

constexpr int Bn = 128;
constexpr int Tn = 4096;
constexpr int Wn = 25;     // per-scale width
constexpr int Sn = 5;      // scales
constexpr int Cn = 125;    // channels (padded to 128 in ybuf)
constexpr int TILE = 256;  // T-tile for stencil kernels

typedef __attribute__((ext_vector_type(16))) __bf16 v16bf;
typedef __attribute__((ext_vector_type(8)))  float  v8f;

union V16U { v16bf v; unsigned short u[16]; };

__device__ __forceinline__ unsigned short f2bf(float f) {
  unsigned int x = __float_as_uint(f);
  x += 0x7fffu + ((x >> 16) & 1u);      // round-to-nearest-even
  return (unsigned short)(x >> 16);
}
__device__ __forceinline__ float bf2f(unsigned short u) {
  return __uint_as_float(((unsigned int)u) << 16);
}

// ---------------------------------------------------------------------------
// accumulator zero init (pooled + avgsum)
__global__ void zero_kernel(float* __restrict__ p, int n) {
  int i = blockIdx.x * blockDim.x + threadIdx.x;
  if (i < n) p[i] = 0.f;
}

// ---------------------------------------------------------------------------
// scale 4: y4 = relu(bn4(conv1x1(lf[8:10]))) -> ybuf[.][.][100..124] (bf16),
// also zeroes padding channels 125..127 of every row.
__global__ __launch_bounds__(TILE) void scale4_kernel(
    const float* __restrict__ lf, const float* __restrict__ w1,
    const float* __restrict__ b1, const float* __restrict__ bng,
    const float* __restrict__ bnb, const float* __restrict__ bnm,
    const float* __restrict__ bnv, unsigned short* __restrict__ ybuf) {
  __shared__ float w1s[Wn][2], b1s[Wn], sc_s[Wn], sh_s[Wn];
  const int b = blockIdx.y;
  const int tid = threadIdx.x;
  const int t = blockIdx.x * TILE + tid;
  if (tid < Wn) {
    w1s[tid][0] = w1[4 * Wn * 2 + tid * 2 + 0];
    w1s[tid][1] = w1[4 * Wn * 2 + tid * 2 + 1];
    b1s[tid] = b1[4 * Wn + tid];
    float sc = bng[4 * Wn + tid] * rsqrtf(bnv[4 * Wn + tid] + EPSV);
    sc_s[tid] = sc;
    sh_s[tid] = bnb[4 * Wn + tid] - bnm[4 * Wn + tid] * sc;
  }
  __syncthreads();
  const float l0 = lf[((size_t)b * 10 + 8) * Tn + t];
  const float l1 = lf[((size_t)b * 10 + 9) * Tn + t];
  const size_t row = ((size_t)b * Tn + t) << 7;  // *128 padded channels
#pragma unroll
  for (int w = 0; w < Wn; ++w) {
    float v = l0 * w1s[w][0] + l1 * w1s[w][1] + b1s[w];
    v = fmaxf(v * sc_s[w] + sh_s[w], 0.f);
    ybuf[row + 4 * Wn + w] = f2bf(v);
  }
  ybuf[row + 125] = 0;  // keep padding deterministic & non-NaN
  ybuf[row + 126] = 0;
  ybuf[row + 127] = 0;
}

// ---------------------------------------------------------------------------
// scale i (3..0): y_i = relu(bn_i(conv3(xs_i + y_{i+1})))
__global__ __launch_bounds__(TILE) void conv_scale_kernel(
    const float* __restrict__ lf, const float* __restrict__ w1,
    const float* __restrict__ b1, const float* __restrict__ w3,
    const float* __restrict__ b3, const float* __restrict__ bng,
    const float* __restrict__ bnb, const float* __restrict__ bnm,
    const float* __restrict__ bnv, unsigned short* __restrict__ ybuf,
    int scale) {
  __shared__ float z[Wn][TILE + 2];       // stencil tile with halo, zero padded
  __shared__ float w3s[Wn * Wn * 3];
  __shared__ float w1s[Wn][2], b1s[Wn], b3s[Wn], sc_s[Wn], sh_s[Wn];
  const int b = blockIdx.y;
  const int t0 = blockIdx.x * TILE;
  const int tid = threadIdx.x;

  for (int idx = tid; idx < Wn * Wn * 3; idx += TILE)
    w3s[idx] = w3[scale * Wn * Wn * 3 + idx];
  if (tid < Wn) {
    w1s[tid][0] = w1[scale * Wn * 2 + tid * 2 + 0];
    w1s[tid][1] = w1[scale * Wn * 2 + tid * 2 + 1];
    b1s[tid] = b1[scale * Wn + tid];
    b3s[tid] = b3[scale * Wn + tid];
    float sc = bng[scale * Wn + tid] * rsqrtf(bnv[scale * Wn + tid] + EPSV);
    sc_s[tid] = sc;
    sh_s[tid] = bnb[scale * Wn + tid] - bnm[scale * Wn + tid] * sc;
  }
  __syncthreads();

  // z[c][j] = xs_i(c, t0-1+j) + y_{i+1}(c, t0-1+j); zero outside [0,T).
  // idx maps c fastest so lanes read contiguous channels of one row.
  for (int idx = tid; idx < Wn * (TILE + 2); idx += TILE) {
    const int c = idx % Wn;
    const int j = idx / Wn;
    const int t = t0 + j - 1;
    float v = 0.f;
    if (t >= 0 && t < Tn) {
      float xs = lf[((size_t)b * 10 + 2 * scale) * Tn + t] * w1s[c][0] +
                 lf[((size_t)b * 10 + 2 * scale + 1) * Tn + t] * w1s[c][1] +
                 b1s[c];
      v = xs + bf2f(ybuf[(((size_t)b * Tn + t) << 7) + (scale + 1) * Wn + c]);
    }
    z[c][j] = v;
  }
  __syncthreads();

  const int t = t0 + tid;
  float out[Wn];
#pragma unroll
  for (int w = 0; w < Wn; ++w) out[w] = b3s[w];
  for (int c = 0; c < Wn; ++c) {
    const float z0 = z[c][tid], z1 = z[c][tid + 1], z2 = z[c][tid + 2];
#pragma unroll
    for (int w = 0; w < Wn; ++w) {
      const float* wp = &w3s[w * Wn * 3 + c * 3];
      out[w] += wp[0] * z0 + wp[1] * z1 + wp[2] * z2;
    }
  }
  const size_t row = (((size_t)b * Tn + t) << 7) + scale * Wn;
#pragma unroll
  for (int w = 0; w < Wn; ++w) {
    float v = fmaxf(out[w] * sc_s[w] + sh_s[w], 0.f);
    ybuf[row + w] = f2bf(v);
  }
}

// ---------------------------------------------------------------------------
// partial temporal sums for gating: avgsum[b*C+c] += sum over a T-chunk
__global__ __launch_bounds__(128) void avg_kernel(
    const unsigned short* __restrict__ ybuf, float* __restrict__ avgsum) {
  const int b = blockIdx.y;
  const int t0 = blockIdx.x * 512;
  const int c = threadIdx.x;
  if (c >= Cn) return;
  float s = 0.f;
  for (int t = t0; t < t0 + 512; ++t)
    s += bf2f(ybuf[(((size_t)b * Tn + t) << 7) + c]);
  atomicAdd(&avgsum[b * Cn + c], s);
}

// ---------------------------------------------------------------------------
// gating MLP: 125 -> 31 (relu) -> 5 (sigmoid);  avg = avgsum / T
__global__ __launch_bounds__(128) void gate_kernel(
    const float* __restrict__ avgsum, const float* __restrict__ gw1,
    const float* __restrict__ gb1, const float* __restrict__ gw2,
    const float* __restrict__ gb2, float* __restrict__ gate) {
  __shared__ float sA[Bn * Cn];
  const int tid = threadIdx.x;
  for (int idx = tid; idx < Bn * Cn; idx += 128)
    sA[idx] = avgsum[idx] * (1.f / Tn);
  __syncthreads();
  const int b = tid;
  float h[31];
#pragma unroll
  for (int j = 0; j < 31; ++j) {
    float s = gb1[j];
    for (int c = 0; c < Cn; ++c) s += sA[b * Cn + c] * gw1[j * Cn + c];
    h[j] = fmaxf(s, 0.f);
  }
#pragma unroll
  for (int s5 = 0; s5 < Sn; ++s5) {
    float g = gb2[s5];
#pragma unroll
    for (int j = 0; j < 31; ++j) g += h[j] * gw2[s5 * 31 + j];
    gate[b * Sn + s5] = 1.f / (1.f + __expf(-g));
  }
}

// ---------------------------------------------------------------------------
// WMMA projection with the gate folded into A:
//   proj[o,t] = sum_c (pf_w[o,c]*gate[b,c/25]) * y[c,t]
// B operand = raw bf16 y rows -> one 32B vector load per lane per k-step.
// Fused epilogue: + pf_b + identity(10ch), BN, ReLU, temporal pooling.
__global__ __launch_bounds__(128) void proj_wmma_kernel(
    const float* __restrict__ lf, const unsigned short* __restrict__ ybuf,
    const float* __restrict__ gate, const float* __restrict__ pfw,
    const float* __restrict__ pfb, const float* __restrict__ idw,
    const float* __restrict__ idb, const float* __restrict__ fg,
    const float* __restrict__ fb, const float* __restrict__ fm,
    const float* __restrict__ fv, float* __restrict__ pooled) {
  __shared__ unsigned short pfw_s[128 * 128];  // bf16 gated A, 125->128 padded
  __shared__ float idw_s[Cn * 10];
  __shared__ float fsc_s[128], fsh_s[128], pfb_s[128], idb_s[128], gch_s[128];
  __shared__ float pooled_s[128];

  const int b = blockIdx.y;
  const int t0 = blockIdx.x * 64;
  const int tid = threadIdx.x;
  const int wave = tid >> 5;
  const int lane = tid & 31;
  const int half = lane >> 4;   // 0: lanes 0-15, 1: lanes 16-31
  const int ln16 = lane & 15;
  const int tw = t0 + wave * 16 + ln16;  // this lane's t for B and C/D tiles

  // per-channel gate factor first (needed while staging A)
  gch_s[tid] = (tid < Cn) ? gate[b * Sn + tid / Wn] : 0.f;
  pooled_s[tid] = 0.f;
  {
    float sc = 0.f, sh = 0.f, pb = 0.f, ib = 0.f;
    if (tid < Cn) {
      sc = fg[tid] * rsqrtf(fv[tid] + EPSV);
      sh = fb[tid] - fm[tid] * sc;
      pb = pfb[tid];
      ib = idb[tid];
    }
    fsc_s[tid] = sc; fsh_s[tid] = sh; pfb_s[tid] = pb; idb_s[tid] = ib;
  }
  for (int idx = tid; idx < Cn * 10; idx += 128) idw_s[idx] = idw[idx];
  __syncthreads();

  // stage A = gate-scaled pf_w (zero-padded 128x128, bf16)
  for (int idx = tid; idx < 128 * 128; idx += 128) {
    const int o = idx >> 7, c = idx & 127;
    float v = (o < Cn && c < Cn) ? pfw[o * Cn + c] * gch_s[c] : 0.f;
    pfw_s[idx] = f2bf(v);
  }
  __syncthreads();

  // per-lane low_freq samples for the fused identity projection (t fixed/lane)
  float lfc[10];
#pragma unroll
  for (int c = 0; c < 10; ++c) lfc[c] = lf[((size_t)b * 10 + c) * Tn + tw];

  v8f acc[8] = {};  // 8 M-tiles x 16x16 f32 accumulators
  const unsigned short* yrow = ybuf + (((size_t)b * Tn + tw) << 7);

#pragma unroll
  for (int ks = 0; ks < 4; ++ks) {
    const int k0 = ks * 32;
    // B operand (32x16 bf16): lanes 0-15 hold K=k0..k0+15, lanes 16-31 +16.
    // Channel-contiguous bf16 layout -> single 32B vector load per lane.
    V16U bv;
    bv.v = *reinterpret_cast<const v16bf*>(yrow + k0 + half * 16);
#pragma unroll
    for (int mt = 0; mt < 8; ++mt) {
      // A operand (16x32 bf16): M = ln16; e0-15 -> K=klo+e (klo = k0 + half*16)
      V16U av;
      const int m = mt * 16 + ln16;
      const int klo = k0 + half * 16;
#pragma unroll
      for (int e = 0; e < 8; ++e) av.u[e] = pfw_s[m * 128 + klo + e];
#pragma unroll
      for (int e = 0; e < 8; ++e) av.u[8 + e] = pfw_s[m * 128 + klo + 8 + e];
      acc[mt] = __builtin_amdgcn_wmma_f32_16x16x32_bf16(
          false, av.v, false, bv.v, (short)0, acc[mt], false, false);
    }
  }

  // fused epilogue: + pf_b + identity, BN, ReLU, reduce over the 16 t's
#pragma unroll
  for (int mt = 0; mt < 8; ++mt) {
#pragma unroll
    for (int r = 0; r < 8; ++r) {
      const int o = mt * 16 + r + half * 8;  // C/D layout: M = r (+8 hi half)
      float outv = 0.f;
      if (o < Cn) {
        float idv = idb_s[o];
#pragma unroll
        for (int c = 0; c < 10; ++c) idv += idw_s[o * 10 + c] * lfc[c];
        float tot = acc[mt][r] + pfb_s[o] + idv;
        outv = fmaxf(tot * fsc_s[o] + fsh_s[o], 0.f);
      }
      float s = outv;  // sum across N (16 lanes of this half)
      s += __shfl_xor(s, 1, 32);
      s += __shfl_xor(s, 2, 32);
      s += __shfl_xor(s, 4, 32);
      s += __shfl_xor(s, 8, 32);
      if (ln16 == 0 && o < Cn) atomicAdd(&pooled_s[o], s);
    }
  }
  __syncthreads();
  if (tid < Cn) atomicAdd(&pooled[b * Cn + tid], pooled_s[tid]);
}

// ---------------------------------------------------------------------------
// head: /T, LayerNorm over channels, fc1+LeakyReLU(0.2), fc2 -> [128,2]
__global__ __launch_bounds__(128) void head_kernel(
    const float* __restrict__ pooled, const float* __restrict__ lng,
    const float* __restrict__ lnb, const float* __restrict__ f1w,
    const float* __restrict__ f1b, const float* __restrict__ f2w,
    const float* __restrict__ f2b, float* __restrict__ out) {
  __shared__ float sP[Bn * Cn];
  const int tid = threadIdx.x;
  for (int idx = tid; idx < Bn * Cn; idx += 128)
    sP[idx] = pooled[idx] * (1.f / Tn);
  __syncthreads();
  const int b = tid;
  float mu = 0.f;
  for (int c = 0; c < Cn; ++c) mu += sP[b * Cn + c];
  mu *= (1.f / Cn);
  float var = 0.f;
  for (int c = 0; c < Cn; ++c) {
    const float d = sP[b * Cn + c] - mu;
    var += d * d;
  }
  var *= (1.f / Cn);
  const float inv = rsqrtf(var + EPSV);
  for (int c = 0; c < Cn; ++c)  // row-private rewrite, no sync needed
    sP[b * Cn + c] = (sP[b * Cn + c] - mu) * inv * lng[c] + lnb[c];
  float h[64];
#pragma unroll
  for (int j = 0; j < 64; ++j) {
    float s = f1b[j];
    for (int c = 0; c < Cn; ++c) s += sP[b * Cn + c] * f1w[j * Cn + c];
    h[j] = (s > 0.f) ? s : 0.2f * s;
  }
#pragma unroll
  for (int k = 0; k < 2; ++k) {
    float s = f2b[k];
#pragma unroll
    for (int j = 0; j < 64; ++j) s += h[j] * f2w[k * 64 + j];
    out[b * 2 + k] = s;
  }
}

// ---------------------------------------------------------------------------
extern "C" void kernel_launch(void* const* d_in, const int* in_sizes, int n_in,
                              void* d_out, int out_size, void* d_ws,
                              size_t ws_size, hipStream_t stream) {
  (void)in_sizes; (void)n_in; (void)out_size; (void)ws_size;
  const float* lf  = (const float*)d_in[0];
  const float* w1  = (const float*)d_in[1];
  const float* b1  = (const float*)d_in[2];
  const float* w3  = (const float*)d_in[3];
  const float* b3  = (const float*)d_in[4];
  const float* bng = (const float*)d_in[5];
  const float* bnb = (const float*)d_in[6];
  const float* bnm = (const float*)d_in[7];
  const float* bnv = (const float*)d_in[8];
  const float* gw1 = (const float*)d_in[9];
  const float* gb1 = (const float*)d_in[10];
  const float* gw2 = (const float*)d_in[11];
  const float* gb2 = (const float*)d_in[12];
  const float* idw = (const float*)d_in[13];
  const float* idb = (const float*)d_in[14];
  const float* pfw = (const float*)d_in[15];
  const float* pfb = (const float*)d_in[16];
  const float* fbg = (const float*)d_in[17];
  const float* fbb = (const float*)d_in[18];
  const float* fbm = (const float*)d_in[19];
  const float* fbv = (const float*)d_in[20];
  const float* lng = (const float*)d_in[21];
  const float* lnb = (const float*)d_in[22];
  const float* f1w = (const float*)d_in[23];
  const float* f1b = (const float*)d_in[24];
  const float* f2w = (const float*)d_in[25];
  const float* f2b = (const float*)d_in[26];

  // workspace layout
  unsigned short* ybuf = (unsigned short*)d_ws;  // [B][T][128] bf16
  float* fbase =
      (float*)((char*)d_ws + (size_t)Bn * Tn * 128 * sizeof(unsigned short));
  float* pooled = fbase;             // [B,125] sums
  float* avgsum = fbase + Bn * Cn;   // [B,125] sums
  float* gateb  = avgsum + Bn * Cn;  // [B,5]

  zero_kernel<<<(2 * Bn * Cn + 255) / 256, 256, 0, stream>>>(pooled,
                                                             2 * Bn * Cn);

  dim3 gTiles(Tn / TILE, Bn);
  scale4_kernel<<<gTiles, TILE, 0, stream>>>(lf, w1, b1, bng, bnb, bnm, bnv,
                                             ybuf);
  for (int s = 3; s >= 0; --s)  // sequential Res2Net recursion
    conv_scale_kernel<<<gTiles, TILE, 0, stream>>>(lf, w1, b1, w3, b3, bng,
                                                   bnb, bnm, bnv, ybuf, s);

  avg_kernel<<<dim3(Tn / 512, Bn), 128, 0, stream>>>(ybuf, avgsum);
  gate_kernel<<<1, 128, 0, stream>>>(avgsum, gw1, gb1, gw2, gb2, gateb);

  dim3 gProj(Tn / 64, Bn);
  proj_wmma_kernel<<<gProj, 128, 0, stream>>>(lf, ybuf, gateb, pfw, pfb, idw,
                                              idb, fbg, fbb, fbm, fbv, pooled);

  head_kernel<<<1, 128, 0, stream>>>(pooled, lng, lnb, f1w, f1b, f2w, f2b,
                                     (float*)d_out);
}
